// ZImageDiT_64776696758873
// MI455X (gfx1250) — matrix operand order repaired and checked by reading, not compiled, not run on hardware
//
#include <hip/hip_runtime.h>
#include <math.h>
#include <stddef.h>

#define S_LEN   2048
#define D_DIM   3072
#define H_CNT   24
#define HD_DIM  128
#define FFN_DIM 8192
#define ADALN_N 256

typedef __attribute__((ext_vector_type(16))) __bf16 v16bf;
typedef __attribute__((ext_vector_type(8)))  __bf16 v8bf;
typedef __attribute__((ext_vector_type(8)))  float  v8f;
typedef __attribute__((ext_vector_type(8)))  unsigned short us8;
typedef __attribute__((ext_vector_type(4)))  float  f4;

// Native bf16 convert (v_cvt on gfx1250) instead of software RNE sequence.
__device__ __forceinline__ unsigned short f2bf(float f) {
    union { __bf16 b; unsigned short u; } cv;
    cv.b = (__bf16)f;
    return cv.u;
}
__device__ __forceinline__ float bf2f(unsigned short h) {
    return __uint_as_float(((unsigned int)h) << 16);
}

// CDNA5 async global->LDS copy (ASYNCcnt-tracked), guarded so either toolchain compiles.
#if __has_builtin(__builtin_amdgcn_global_load_async_to_lds_b128) && \
    __has_builtin(__builtin_amdgcn_s_wait_asynccnt)
#define HAVE_ASYNC_LDS 1
typedef int v4i_vs __attribute__((vector_size(16)));
typedef __attribute__((address_space(1))) v4i_vs* g_b128_ptr;
typedef __attribute__((address_space(3))) v4i_vs* l_b128_ptr;
__device__ __forceinline__ void async_copy_b128(const unsigned short* g, unsigned short* l) {
    __builtin_amdgcn_global_load_async_to_lds_b128((g_b128_ptr)(g), (l_b128_ptr)(l), 0, 0);
}
#endif

// ---------------------------------------------------------------- adaLN mod
// mod[j] for j in [0,4D): seg0/2 -> 1+m (scales), seg1/3 -> tanh(m) (gates)
__global__ __launch_bounds__(256) void adaln_kernel(
    const float* __restrict__ ain, const float* __restrict__ W,
    const float* __restrict__ b, float* __restrict__ mod)
{
    int j = blockIdx.x * 256 + threadIdx.x;     // < 4*D
    float acc = b[j];
    for (int i = 0; i < ADALN_N; ++i)
        acc = fmaf(ain[i], W[(size_t)i * (4 * D_DIM) + j], acc);
    int seg = j / D_DIM;
    mod[j] = (seg == 0 || seg == 2) ? (1.0f + acc) : tanhf(acc);
}

// ---------------------------------------------------------------- RoPE tables
__global__ void rope_kernel(const int* __restrict__ ids,
                            float* __restrict__ Cos, float* __restrict__ Sin)
{
    int s = blockIdx.x, t = threadIdx.x;        // t in [0,64)
    int axis, i, d;
    if (t < 8)       { axis = 0; i = t;      d = 16; }
    else if (t < 36) { axis = 1; i = t - 8;  d = 56; }
    else             { axis = 2; i = t - 36; d = 56; }
    float freq = powf(256.0f, -(2.0f * (float)i) / (float)d);
    float ang  = (float)ids[s * 3 + axis] * freq;
    Cos[s * 64 + t] = cosf(ang);
    Sin[s * 64 + t] = sinf(ang);
}

// ------------------------------------------------- rmsnorm * scale -> bf16
__global__ __launch_bounds__(256) void rmsnorm_scale_kernel(
    const float* __restrict__ X, const float* __restrict__ w,
    const float* __restrict__ scale, unsigned short* __restrict__ Out,
    float eps)
{
    int row = blockIdx.x, tid = threadIdx.x;
    int lane = tid & 31, wid = tid >> 5;
    __shared__ float red[8];
    __shared__ float sInv;
    const float* xp = X + (size_t)row * D_DIM;
    float ss = 0.f;
    for (int i = tid; i < D_DIM; i += 256) { float v = xp[i]; ss += v * v; }
    for (int m = 1; m < 32; m <<= 1) ss += __shfl_xor(ss, m, 32);
    if (lane == 0) red[wid] = ss;
    __syncthreads();
    if (tid == 0) {
        float t = 0.f;
        for (int i = 0; i < 8; ++i) t += red[i];
        sInv = rsqrtf(t / (float)D_DIM + eps);
    }
    __syncthreads();
    float inv = sInv;
    unsigned short* op = Out + (size_t)row * D_DIM;
    for (int i = tid; i < D_DIM; i += 256)
        op[i] = f2bf(xp[i] * inv * w[i] * scale[i]);
}

// ----------------------------------------- x + gate * rmsnorm(y, w) (f32)
__global__ __launch_bounds__(256) void residual_gate_norm_kernel(
    const float* __restrict__ Xin, const float* __restrict__ Y,
    const float* __restrict__ w, const float* __restrict__ gate,
    float* __restrict__ Xout)
{
    int row = blockIdx.x, tid = threadIdx.x;
    int lane = tid & 31, wid = tid >> 5;
    __shared__ float red[8];
    __shared__ float sInv;
    const float* yp = Y + (size_t)row * D_DIM;
    const float* xp = Xin + (size_t)row * D_DIM;
    float ss = 0.f;
    for (int i = tid; i < D_DIM; i += 256) { float v = yp[i]; ss += v * v; }
    for (int m = 1; m < 32; m <<= 1) ss += __shfl_xor(ss, m, 32);
    if (lane == 0) red[wid] = ss;
    __syncthreads();
    if (tid == 0) {
        float t = 0.f;
        for (int i = 0; i < 8; ++i) t += red[i];
        sInv = rsqrtf(t / (float)D_DIM + 1e-6f);
    }
    __syncthreads();
    float inv = sInv;
    float* op = Xout + (size_t)row * D_DIM;
    for (int i = tid; i < D_DIM; i += 256)
        op[i] = xp[i] + gate[i] * (yp[i] * inv * w[i]);
}

// --------------------------------- per-head rmsnorm + rope, f32 -> bf16 [H][S][HD]
__global__ __launch_bounds__(256) void qk_norm_rope_kernel(
    const float* __restrict__ Q, const float* __restrict__ w,
    const float* __restrict__ Cos, const float* __restrict__ Sin,
    unsigned short* __restrict__ Oh)
{
    int lane = threadIdx.x & 31, wid = threadIdx.x >> 5;
    int idx = blockIdx.x * 8 + wid;             // < S*H
    int s = idx / H_CNT, h = idx % H_CNT;
    const float* qp = Q + (size_t)s * D_DIM + h * HD_DIM;
    int d0 = 4 * lane;
    float x0 = qp[d0], x1 = qp[d0 + 1], x2 = qp[d0 + 2], x3 = qp[d0 + 3];
    float ss = x0 * x0 + x1 * x1 + x2 * x2 + x3 * x3;
    for (int m = 1; m < 32; m <<= 1) ss += __shfl_xor(ss, m, 32);
    float inv = rsqrtf(ss / (float)HD_DIM + 1e-5f);
    x0 *= inv * w[d0];     x1 *= inv * w[d0 + 1];
    x2 *= inv * w[d0 + 2]; x3 *= inv * w[d0 + 3];
    // pairs (d0,d0+1) -> angle 2*lane ; (d0+2,d0+3) -> angle 2*lane+1
    float c0 = Cos[s * 64 + 2 * lane],     s0 = Sin[s * 64 + 2 * lane];
    float c1 = Cos[s * 64 + 2 * lane + 1], s1 = Sin[s * 64 + 2 * lane + 1];
    float r0 = x0 * c0 - x1 * s0, i0 = x0 * s0 + x1 * c0;
    float r1 = x2 * c1 - x3 * s1, i1 = x2 * s1 + x3 * c1;
    unsigned short* op = Oh + ((size_t)h * S_LEN + s) * HD_DIM + d0;
    op[0] = f2bf(r0); op[1] = f2bf(i0); op[2] = f2bf(r1); op[3] = f2bf(i1);
}

// ----------------------------- V: f32 [S][H*HD] -> bf16 [H][HD][S] (coalesced out)
__global__ void v_transpose_kernel(const float* __restrict__ Vf,
                                   unsigned short* __restrict__ Vt)
{
    size_t id = (size_t)blockIdx.x * 256 + threadIdx.x;   // < H*HD*S
    int s = (int)(id % S_LEN);
    size_t hd = id / S_LEN;
    Vt[id] = f2bf(Vf[(size_t)s * D_DIM + hd]);
}

// ---------------------------------------------------------------- silu(u)*g
__global__ void silu_mul_kernel(const unsigned short* __restrict__ U,
                                const unsigned short* __restrict__ G,
                                unsigned short* __restrict__ Hh)
{
    size_t id = (size_t)blockIdx.x * 256 + threadIdx.x;
    float u = bf2f(U[id]);
    float g = bf2f(G[id]);
    float si = u / (1.0f + __expf(-u));
    Hh[id] = f2bf(si * g);
}

// ================================================================= GEMM
// C[M,N] = A_bf16[M,K] @ W_f32[K,N]  (W converted to bf16 at LDS store)
// block tile 128x128, BK=32, 8 waves: wave = 4 m-tiles x 2 n-tiles of 16x16
__global__ __launch_bounds__(256) void gemm_bf16_kernel(
    const unsigned short* __restrict__ A, const float* __restrict__ B,
    float* __restrict__ Cf, unsigned short* __restrict__ Cb,
    int M, int N, int K, int outBf16)
{
    __shared__ __align__(16) unsigned short As[128][40]; // [m][k]
    __shared__ __align__(16) unsigned short Bs[128][40]; // [n][k] (transposed)
    const int tid  = threadIdx.x;
    const int lane = tid & 31;
    const int wid  = tid >> 5;
    const int half = lane >> 4, ln = lane & 15;
    const int wm = wid & 1, wn = wid >> 1;
    const int m0 = blockIdx.y * 128, n0 = blockIdx.x * 128;

    v8f acc[4][2];
    for (int i = 0; i < 4; ++i)
        for (int j = 0; j < 2; ++j)
            acc[i][j] = (v8f){0, 0, 0, 0, 0, 0, 0, 0};

    for (int k0 = 0; k0 < K; k0 += 32) {
        // A tile: 128x32 bf16, 16 elems/thread (async direct-to-LDS when available)
#ifdef HAVE_ASYNC_LDS
        for (int i = 0; i < 2; ++i) {
            int c = tid + 256 * i;
            int row = c >> 2, koff = (c & 3) * 8;
            async_copy_b128(A + (size_t)(m0 + row) * K + k0 + koff, &As[row][koff]);
        }
#else
        for (int i = 0; i < 2; ++i) {
            int c = tid + 256 * i;
            int row = c >> 2, koff = (c & 3) * 8;
            us8 v = *(const us8*)(A + (size_t)(m0 + row) * K + k0 + koff);
            *(us8*)&As[row][koff] = v;
        }
#endif
        // B tile: 32x128 f32 -> bf16, stored transposed [n][k].
        // Issue all 4 global loads first (keep 4 VMEM loads in flight),
        // then convert + transpose-store.
        {
            f4 v[4];
            int kk[4], nf[4];
            for (int i = 0; i < 4; ++i) {
                int c = tid + 256 * i;
                kk[i] = c >> 5;
                nf[i] = (c & 31) * 4;
                v[i] = *(const f4*)(B + (size_t)(k0 + kk[i]) * N + n0 + nf[i]);
            }
            for (int i = 0; i < 4; ++i) {
                Bs[nf[i] + 0][kk[i]] = f2bf(v[i][0]);
                Bs[nf[i] + 1][kk[i]] = f2bf(v[i][1]);
                Bs[nf[i] + 2][kk[i]] = f2bf(v[i][2]);
                Bs[nf[i] + 3][kk[i]] = f2bf(v[i][3]);
            }
        }
        if (k0 + 32 < K) {
            __builtin_prefetch(A + (size_t)(m0 + (tid >> 1)) * K + k0 + 32 + (tid & 1) * 16, 0, 1);
            __builtin_prefetch(B + (size_t)(k0 + 32 + (tid >> 3)) * N + n0 + (tid & 7) * 16, 0, 1);
        }
#ifdef HAVE_ASYNC_LDS
        __builtin_amdgcn_s_wait_asynccnt(0);   // my async tile data is in LDS
#endif
        __syncthreads();

        v16bf af[4];
        for (int mt = 0; mt < 4; ++mt) {
            int row = wm * 64 + mt * 16 + ln;
            ((v8bf*)&af[mt])[0] = *(const v8bf*)&As[row][8 * half];
            ((v8bf*)&af[mt])[1] = *(const v8bf*)&As[row][16 + 8 * half];
        }
        for (int nt = 0; nt < 2; ++nt) {
            int col = wn * 32 + nt * 16 + ln;
            v16bf bfv;
            ((v8bf*)&bfv)[0] = *(const v8bf*)&Bs[col][8 * half];
            ((v8bf*)&bfv)[1] = *(const v8bf*)&Bs[col][16 + 8 * half];
            for (int mt = 0; mt < 4; ++mt)
                acc[mt][nt] = __builtin_amdgcn_wmma_f32_16x16x32_bf16(
                    false, af[mt], false, bfv, (short)0, acc[mt][nt], false, false);
        }
        __syncthreads();
    }

    for (int mt = 0; mt < 4; ++mt)
        for (int nt = 0; nt < 2; ++nt)
            for (int r = 0; r < 8; ++r) {
                int grow = m0 + wm * 64 + mt * 16 + r + 8 * half;
                int gcol = n0 + wn * 32 + nt * 16 + ln;
                float v = acc[mt][nt][r];
                if (outBf16) Cb[(size_t)grow * N + gcol] = f2bf(v);
                else         Cf[(size_t)grow * N + gcol] = v;
            }
}

// ================================================================= Flash attn
// one wave = 16 query rows of one head; Q [H][S][HD] bf16, K same, V [H][HD][S] bf16
__global__ __launch_bounds__(256) void flash_attn_kernel(
    const unsigned short* __restrict__ Qh, const unsigned short* __restrict__ Kh,
    const unsigned short* __restrict__ Vt, const unsigned char* __restrict__ mask,
    unsigned short* __restrict__ O)
{
    __shared__ __align__(16) unsigned short Ps[8][16][40];  // per-wave P tile
    const int tid  = threadIdx.x;
    const int lane = tid & 31;
    const int wid  = tid >> 5;
    const int half = lane >> 4, ln = lane & 15;
    const int tilesPerHead = S_LEN / 128;
    const int head = blockIdx.x / tilesPerHead;
    const int tile = blockIdx.x % tilesPerHead;
    const int q0 = (tile * 8 + wid) * 16;
    const float scale = 0.088388347648318447f;   // 1/sqrt(128)

    // Q fragments (A-layout), kept in VGPRs across whole KV loop
    v16bf qa[4];
    const unsigned short* Qp = Qh + ((size_t)head * S_LEN + q0 + ln) * HD_DIM;
    for (int t = 0; t < 4; ++t) {
        const unsigned short* p = Qp + 32 * t;
        ((v8bf*)&qa[t])[0] = *(const v8bf*)(p + 8 * half);
        ((v8bf*)&qa[t])[1] = *(const v8bf*)(p + 16 + 8 * half);
    }

    float run_max[8], run_sum[8];
    v8f o[8];
    for (int r = 0; r < 8; ++r) { run_max[r] = -1e30f; run_sum[r] = 0.f; }
    for (int j = 0; j < 8; ++j) o[j] = (v8f){0, 0, 0, 0, 0, 0, 0, 0};

    for (int k0 = 0; k0 < S_LEN; k0 += 32) {
        // ---- scores: two 16x16 tiles (keys k0..k0+15, k0+16..k0+31)
        v8f s01[2];
        float mb[2];
        for (int tt = 0; tt < 2; ++tt) {
            int key = k0 + tt * 16 + ln;
            mb[tt] = mask[key] ? 0.f : -1e30f;
            const unsigned short* kp =
                Kh + ((size_t)head * S_LEN + key) * HD_DIM + 8 * half;
            v8f cc = (v8f){0, 0, 0, 0, 0, 0, 0, 0};
            for (int t = 0; t < 4; ++t) {
                v16bf kb;
                ((v8bf*)&kb)[0] = *(const v8bf*)(kp + 32 * t);
                ((v8bf*)&kb)[1] = *(const v8bf*)(kp + 32 * t + 16);
                cc = __builtin_amdgcn_wmma_f32_16x16x32_bf16(
                    false, qa[t], false, kb, (short)0, cc, false, false);
            }
            s01[tt] = cc;
        }
        // ---- online softmax (rows r+8*half; columns striped over 16 lanes)
        for (int r = 0; r < 8; ++r) {
            float a = s01[0][r] * scale + mb[0];
            float b = s01[1][r] * scale + mb[1];
            float mx = fmaxf(a, b);
            mx = fmaxf(mx, __shfl_xor(mx, 1, 32));
            mx = fmaxf(mx, __shfl_xor(mx, 2, 32));
            mx = fmaxf(mx, __shfl_xor(mx, 4, 32));
            mx = fmaxf(mx, __shfl_xor(mx, 8, 32));
            float nm   = fmaxf(run_max[r], mx);
            float corr = __expf(run_max[r] - nm);
            float p0 = __expf(a - nm);
            float p1 = __expf(b - nm);
            float ls = p0 + p1;
            ls += __shfl_xor(ls, 1, 32);
            ls += __shfl_xor(ls, 2, 32);
            ls += __shfl_xor(ls, 4, 32);
            ls += __shfl_xor(ls, 8, 32);
            run_sum[r] = run_sum[r] * corr + ls;
            run_max[r] = nm;
            for (int j = 0; j < 8; ++j) o[j][r] *= corr;
            Ps[wid][r + 8 * half][ln]      = f2bf(p0);  // C-layout -> LDS
            Ps[wid][r + 8 * half][16 + ln] = f2bf(p1);
        }
        // ---- P (16x32) as A-fragment from LDS (in-wave DS ordering)
        v16bf pa;
        ((v8bf*)&pa)[0] = *(const v8bf*)&Ps[wid][ln][8 * half];
        ((v8bf*)&pa)[1] = *(const v8bf*)&Ps[wid][ln][16 + 8 * half];
        // ---- o += P @ V   (V^T rows are contiguous B-fragments)
        for (int j = 0; j < 8; ++j) {
            const unsigned short* vp =
                Vt + ((size_t)head * HD_DIM + j * 16 + ln) * S_LEN + k0 + 8 * half;
            v16bf vb;
            ((v8bf*)&vb)[0] = *(const v8bf*)vp;
            ((v8bf*)&vb)[1] = *(const v8bf*)(vp + 16);
            o[j] = __builtin_amdgcn_wmma_f32_16x16x32_bf16(
                false, pa, false, vb, (short)0, o[j], false, false);
        }
    }
    // ---- write O (bf16, token-major [S][D])
    for (int j = 0; j < 8; ++j)
        for (int r = 0; r < 8; ++r) {
            float v = o[j][r] / run_sum[r];
            O[(size_t)(q0 + r + 8 * half) * D_DIM + head * HD_DIM + j * 16 + ln] = f2bf(v);
        }
}

// ================================================================= host
static inline size_t alignup256(size_t x) { return (x + 255) & ~(size_t)255; }

extern "C" void kernel_launch(void* const* d_in, const int* in_sizes, int n_in,
                              void* d_out, int out_size, void* d_ws, size_t ws_size,
                              hipStream_t stream)
{
    (void)in_sizes; (void)n_in; (void)out_size; (void)ws_size;
    const float* x            = (const float*)d_in[0];
    const float* adaln_input  = (const float*)d_in[1];
    const unsigned char* attn_mask = (const unsigned char*)d_in[2];
    const int*   ids          = (const int*)d_in[3];
    const float* attn_norm1_w = (const float*)d_in[4];
    const float* attn_norm2_w = (const float*)d_in[5];
    const float* ffn_norm1_w  = (const float*)d_in[6];
    const float* ffn_norm2_w  = (const float*)d_in[7];
    const float* norm_q_w     = (const float*)d_in[8];
    const float* norm_k_w     = (const float*)d_in[9];
    const float* wq           = (const float*)d_in[10];
    const float* wk           = (const float*)d_in[11];
    const float* wv           = (const float*)d_in[12];
    const float* wo           = (const float*)d_in[13];
    const float* w1           = (const float*)d_in[14];
    const float* w2           = (const float*)d_in[15];
    const float* w3           = (const float*)d_in[16];
    const float* adaln_w      = (const float*)d_in[17];
    const float* adaln_b      = (const float*)d_in[18];
    float* out = (float*)d_out;

    char* base = (char*)d_ws;
    size_t off = 0;
    auto alloc = [&](size_t bytes) -> char* {
        char* r = base + off;
        off = alignup256(off + bytes);
        return r;
    };
    float*          mod    = (float*)alloc((size_t)4 * D_DIM * 4);
    float*          cosb   = (float*)alloc((size_t)S_LEN * 64 * 4);
    float*          sinb   = (float*)alloc((size_t)S_LEN * 64 * 4);
    unsigned short* xs     = (unsigned short*)alloc((size_t)S_LEN * D_DIM * 2);
    float*          qf     = (float*)alloc((size_t)S_LEN * D_DIM * 4);
    float*          kf     = (float*)alloc((size_t)S_LEN * D_DIM * 4);
    float*          vf     = (float*)alloc((size_t)S_LEN * D_DIM * 4);
    unsigned short* Qhb    = (unsigned short*)alloc((size_t)S_LEN * D_DIM * 2);
    unsigned short* Khb    = (unsigned short*)alloc((size_t)S_LEN * D_DIM * 2);
    unsigned short* Vtb    = (unsigned short*)alloc((size_t)S_LEN * D_DIM * 2);
    unsigned short* attnb  = (unsigned short*)alloc((size_t)S_LEN * D_DIM * 2);
    float*          attnof = (float*)alloc((size_t)S_LEN * D_DIM * 4);
    float*          x2     = (float*)alloc((size_t)S_LEN * D_DIM * 4);
    unsigned short* xfb    = (unsigned short*)alloc((size_t)S_LEN * D_DIM * 2);
    unsigned short* ub     = (unsigned short*)alloc((size_t)S_LEN * FFN_DIM * 2);
    unsigned short* gb     = (unsigned short*)alloc((size_t)S_LEN * FFN_DIM * 2);
    unsigned short* hb     = (unsigned short*)alloc((size_t)S_LEN * FFN_DIM * 2);
    float*          ffnf   = (float*)alloc((size_t)S_LEN * D_DIM * 4);

    const float* scale_msa = mod;
    const float* gate_msa  = mod + D_DIM;
    const float* scale_mlp = mod + 2 * D_DIM;
    const float* gate_mlp  = mod + 3 * D_DIM;

    adaln_kernel<<<(4 * D_DIM) / 256, 256, 0, stream>>>(adaln_input, adaln_w, adaln_b, mod);
    rope_kernel<<<S_LEN, 64, 0, stream>>>(ids, cosb, sinb);
    rmsnorm_scale_kernel<<<S_LEN, 256, 0, stream>>>(x, attn_norm1_w, scale_msa, xs, 1e-6f);

    dim3 gDD(D_DIM / 128, S_LEN / 128);
    gemm_bf16_kernel<<<gDD, 256, 0, stream>>>(xs, wq, qf, nullptr, S_LEN, D_DIM, D_DIM, 0);
    gemm_bf16_kernel<<<gDD, 256, 0, stream>>>(xs, wk, kf, nullptr, S_LEN, D_DIM, D_DIM, 0);
    gemm_bf16_kernel<<<gDD, 256, 0, stream>>>(xs, wv, vf, nullptr, S_LEN, D_DIM, D_DIM, 0);

    qk_norm_rope_kernel<<<(S_LEN * H_CNT) / 8, 256, 0, stream>>>(qf, norm_q_w, cosb, sinb, Qhb);
    qk_norm_rope_kernel<<<(S_LEN * H_CNT) / 8, 256, 0, stream>>>(kf, norm_k_w, cosb, sinb, Khb);
    v_transpose_kernel<<<(S_LEN * D_DIM) / 256, 256, 0, stream>>>(vf, Vtb);

    flash_attn_kernel<<<H_CNT * (S_LEN / 128), 256, 0, stream>>>(Qhb, Khb, Vtb, attn_mask, attnb);

    gemm_bf16_kernel<<<gDD, 256, 0, stream>>>(attnb, wo, attnof, nullptr, S_LEN, D_DIM, D_DIM, 0);
    residual_gate_norm_kernel<<<S_LEN, 256, 0, stream>>>(x, attnof, attn_norm2_w, gate_msa, x2);

    rmsnorm_scale_kernel<<<S_LEN, 256, 0, stream>>>(x2, ffn_norm1_w, scale_mlp, xfb, 1e-6f);

    dim3 gDF(FFN_DIM / 128, S_LEN / 128);
    gemm_bf16_kernel<<<gDF, 256, 0, stream>>>(xfb, w1, nullptr, ub, S_LEN, FFN_DIM, D_DIM, 1);
    gemm_bf16_kernel<<<gDF, 256, 0, stream>>>(xfb, w3, nullptr, gb, S_LEN, FFN_DIM, D_DIM, 1);
    silu_mul_kernel<<<(S_LEN * FFN_DIM) / 256, 256, 0, stream>>>(ub, gb, hb);

    gemm_bf16_kernel<<<gDD, 256, 0, stream>>>(hb, w2, ffnf, nullptr, S_LEN, D_DIM, FFN_DIM, 0);
    residual_gate_norm_kernel<<<S_LEN, 256, 0, stream>>>(x2, ffnf, ffn_norm2_w, gate_mlp, out);
}
